// WindowAttention_3865470566916
// MI455X (gfx1250) — matrix operand order repaired and checked by reading, not compiled, not run on hardware
//
#include <hip/hip_runtime.h>

typedef _Float16 f16;
typedef __attribute__((ext_vector_type(16))) _Float16 v16h;
typedef __attribute__((ext_vector_type(8)))  float   v8f;

#define NH   6
#define NTOK 256
#define HD   32
#define NWIN 16

union ABReg { v16h h; unsigned u[8]; };
union CFrag { uint4 u4; f16 f[8]; };

// ---------------------------------------------------------------------------
// Kernel 1: CPB MLP  ->  biasTab[h][pos] = 16*sigmoid( relu(table@w1.T+b1) @ w2.T )
// ---------------------------------------------------------------------------
__global__ __launch_bounds__(256)
void cpb_bias_kernel(const float* __restrict__ table,
                     const float* __restrict__ w1,
                     const float* __restrict__ b1,
                     const float* __restrict__ w2,
                     float* __restrict__ biasTab)
{
    int tid = blockIdx.x * 256 + threadIdx.x;
    if (tid >= 961 * NH) return;
    int pos = tid / NH;
    int h   = tid % NH;
    float t0 = table[pos * 2 + 0];
    float t1 = table[pos * 2 + 1];
    const float* w2h = w2 + h * 512;
    float acc = 0.f;
    for (int j = 0; j < 512; ++j) {
        float hid = fmaf(t0, w1[2 * j + 0], fmaf(t1, w1[2 * j + 1], b1[j]));
        hid = fmaxf(hid, 0.f);
        acc = fmaf(hid, w2h[j], acc);
    }
    biasTab[h * 961 + pos] = 16.f / (1.f + __expf(-acc));
}

// ---------------------------------------------------------------------------
// Kernel 2: fuse bias gather + shift mask into f16 table pre-swizzled into the
// WMMA C/D lane layout: fused[h][win][tile_r][tile_c][half][lm][v], so a lane
// grabs its whole 16x16-tile C fragment with one 16B load.
// element (r,c): r = tile_r*16 + v + 8*half, c = tile_c*16 + lm
// ---------------------------------------------------------------------------
__global__ __launch_bounds__(256)
void fuse_bias_mask_kernel(const int*   __restrict__ index,
                           const float* __restrict__ maskg,
                           const float* __restrict__ biasTab,
                           f16* __restrict__ fused)
{
    int gid = blockIdx.x * 256 + threadIdx.x;          // NH*16*256*256 total
    if (gid >= NH * NWIN * NTOK * NTOK) return;
    int v    = gid & 7;
    int lm   = (gid >> 3) & 15;
    int half = (gid >> 7) & 1;
    int tc   = (gid >> 8) & 15;
    int tr   = (gid >> 12) & 15;
    int win  = (gid >> 16) & 15;
    int h    = gid >> 20;
    int r = tr * 16 + v + 8 * half;
    int c = tc * 16 + lm;
    float val = biasTab[h * 961 + index[r * NTOK + c]]
              + maskg[win * NTOK * NTOK + r * NTOK + c];
    fused[gid] = (f16)val;
}

// ---------------------------------------------------------------------------
// Kernel 3: one block per (window, head). 256 threads = 8 wave32.
// FUSED=true  -> C operand streamed from the fused f16 table (1 b128/tile/lane)
// FUSED=false -> 3-way gather fallback (if workspace too small)
// ---------------------------------------------------------------------------
template<bool FUSED>
__global__ __launch_bounds__(256)
void swin_attn_kernel(const float* __restrict__ qkv,
                      const float* __restrict__ maskg,
                      const float* __restrict__ logit_scale,
                      const int*   __restrict__ index,
                      const float* __restrict__ biasTab,
                      const f16*   __restrict__ fused,
                      float* __restrict__ out)
{
    extern __shared__ char smem[];
    f16* Qn = (f16*)smem;                 // [256][32]  normalized*scale, f16
    f16* Kn = Qn + NTOK * HD;             // [256][32]  normalized, f16
    f16* Vt = Kn + NTOK * HD;             // [32][256]  V transposed, f16
    f16* P  = Vt + HD * NTOK;             // 8 waves x [16][256] f16 slabs

    const int blk  = blockIdx.x;
    const int h    = blk % NH;
    const int bw   = blk / NH;
    const int b    = bw >> 4;             // image in batch
    const int wwin = bw & 15;             // window within image (== mask idx)
    const int wy   = wwin >> 2, wx = wwin & 3;

    const int tid = threadIdx.x;
    const float qscale = __expf(fminf(logit_scale[h], 4.605170186f)); // ln(100)

    // ---- load + normalize Q/K, transpose V into LDS (shift folded in) ----
    {
        const int t  = tid;               // token 0..255
        const int iy = t >> 4, ix = t & 15;
        const int y  = (wy * 16 + iy + 8) & 63;
        const int x  = (wx * 16 + ix + 8) & 63;
        const size_t rowbase = ((size_t)b * 4096 + (size_t)y * 64 + x) * 576;

        const float4* qs = (const float4*)(qkv + rowbase +   0 + h * HD);
        const float4* ks = (const float4*)(qkv + rowbase + 192 + h * HD);
        const float4* vs = (const float4*)(qkv + rowbase + 384 + h * HD);
        float4 qr[8], kr[8];
        float qss = 0.f, kss = 0.f;
#pragma unroll
        for (int i = 0; i < 8; ++i) {
            qr[i] = qs[i]; kr[i] = ks[i];
            qss += qr[i].x*qr[i].x + qr[i].y*qr[i].y + qr[i].z*qr[i].z + qr[i].w*qr[i].w;
            kss += kr[i].x*kr[i].x + kr[i].y*kr[i].y + kr[i].z*kr[i].z + kr[i].w*kr[i].w;
        }
        const float qi = qscale / fmaxf(sqrtf(qss), 1e-12f);
        const float ki = 1.0f   / fmaxf(sqrtf(kss), 1e-12f);
#pragma unroll
        for (int i = 0; i < 8; ++i) {
            Qn[t*HD + 4*i+0] = (f16)(qr[i].x * qi);
            Qn[t*HD + 4*i+1] = (f16)(qr[i].y * qi);
            Qn[t*HD + 4*i+2] = (f16)(qr[i].z * qi);
            Qn[t*HD + 4*i+3] = (f16)(qr[i].w * qi);
            Kn[t*HD + 4*i+0] = (f16)(kr[i].x * ki);
            Kn[t*HD + 4*i+1] = (f16)(kr[i].y * ki);
            Kn[t*HD + 4*i+2] = (f16)(kr[i].z * ki);
            Kn[t*HD + 4*i+3] = (f16)(kr[i].w * ki);
            float4 vv = vs[i];
            Vt[(4*i+0)*NTOK + t] = (f16)vv.x;
            Vt[(4*i+1)*NTOK + t] = (f16)vv.y;
            Vt[(4*i+2)*NTOK + t] = (f16)vv.z;
            Vt[(4*i+3)*NTOK + t] = (f16)vv.w;
        }
    }
    __syncthreads();

    const int wave = tid >> 5;
    const int lane = tid & 31;
    const int half = lane >> 4;
    const int lm   = lane & 15;
    const unsigned* Qw = (const unsigned*)Qn;
    const unsigned* Kw = (const unsigned*)Kn;
    const unsigned* Vw = (const unsigned*)Vt;
    f16* Pme = P + wave * 16 * NTOK;      // wave-private P slab
    const unsigned* Pw = (const unsigned*)Pme;
    const float* maskw  = maskg + (size_t)wwin * NTOK * NTOK;
    const float* btab   = biasTab + h * 961;
    const f16*   fusedw = fused + ((size_t)(h * NWIN + wwin) << 16);

    for (int rb = wave; rb < 16; rb += 8) {
        const int r0 = rb * 16;

        // A operand: Q rows r0..r0+15, full K-depth 32.
        ABReg a;
        {
            const int rowd = (r0 + lm) * (HD / 2);   // 16 dwords per row
#pragma unroll
            for (int v = 0; v < 4; ++v) a.u[v]     = Qw[rowd + 4*half + v];
#pragma unroll
            for (int v = 0; v < 4; ++v) a.u[4 + v] = Qw[rowd + 8 + 4*half + v];
        }

        // ---- S = (scale*qn).kn^T + (16*sigmoid(cpb)+mask) via WMMA C ----
        v8f acc[16];
#pragma unroll
        for (int tj = 0; tj < 16; ++tj) {
            v8f c;
            if (FUSED) {
                CFrag cf;
                cf.u4 = *(const uint4*)(fusedw + (((rb * 16 + tj) << 8)
                                                 + (half << 7) + (lm << 3)));
#pragma unroll
                for (int v = 0; v < 8; ++v) c[v] = (float)cf.f[v];
            } else {
#pragma unroll
                for (int v = 0; v < 8; ++v) {
                    const int r   = r0 + v + 8 * half;
                    const int cix = tj * 16 + lm;
                    c[v] = btab[index[r * NTOK + cix]] + maskw[r * NTOK + cix];
                }
            }
            ABReg bm;
            const int rowd = (tj * 16 + lm) * (HD / 2);
#pragma unroll
            for (int v = 0; v < 4; ++v) bm.u[v] = Kw[rowd + 4*half + v];
#pragma unroll
            for (int v = 0; v < 4; ++v) bm.u[4 + v] = Kw[rowd + 8 + 4*half + v];
            acc[tj] = __builtin_amdgcn_wmma_f32_16x16x32_f16(
                false, a.h, false, bm.h, (short)0, c, false, false);
        }

        // ---- softmax over 256 columns ----
        float rmax[8], rsum[8];
#pragma unroll
        for (int v = 0; v < 8; ++v) rmax[v] = -3.0e38f;
#pragma unroll
        for (int tj = 0; tj < 16; ++tj)
#pragma unroll
            for (int v = 0; v < 8; ++v) rmax[v] = fmaxf(rmax[v], acc[tj][v]);
#pragma unroll
        for (int v = 0; v < 8; ++v) {
            float m = rmax[v];
            m = fmaxf(m, __shfl_xor(m, 1, 16));
            m = fmaxf(m, __shfl_xor(m, 2, 16));
            m = fmaxf(m, __shfl_xor(m, 4, 16));
            m = fmaxf(m, __shfl_xor(m, 8, 16));
            rmax[v] = m; rsum[v] = 0.f;
        }
#pragma unroll
        for (int tj = 0; tj < 16; ++tj)
#pragma unroll
            for (int v = 0; v < 8; ++v) {
                float e = __expf(acc[tj][v] - rmax[v]);
                acc[tj][v] = e; rsum[v] += e;
            }
#pragma unroll
        for (int v = 0; v < 8; ++v) {
            float s = rsum[v];
            s += __shfl_xor(s, 1, 16);
            s += __shfl_xor(s, 2, 16);
            s += __shfl_xor(s, 4, 16);
            s += __shfl_xor(s, 8, 16);
            rsum[v] = 1.0f / s;
        }

        // ---- stage P (f16, [16][256]) into wave-private slab ----
#pragma unroll
        for (int tj = 0; tj < 16; ++tj)
#pragma unroll
            for (int v = 0; v < 8; ++v)
                Pme[(v + 8*half) * NTOK + tj * 16 + lm] = (f16)(acc[tj][v] * rsum[v]);

        // ---- O = P x V (16x32 = two 16x16 tiles), K in 8 chunks of 32 ----
        v8f o0 = {}; v8f o1 = {};
#pragma unroll
        for (int kc = 0; kc < 8; ++kc) {
            ABReg ap;
            const int prow = lm * (NTOK / 2) + kc * 16;
#pragma unroll
            for (int v = 0; v < 4; ++v) ap.u[v]     = Pw[prow + 4*half + v];
#pragma unroll
            for (int v = 0; v < 4; ++v) ap.u[4 + v] = Pw[prow + 8 + 4*half + v];
            ABReg b0, b1;
#pragma unroll
            for (int v = 0; v < 4; ++v) {
                b0.u[v]     = Vw[(lm     ) * (NTOK / 2) + kc * 16 + 4*half + v];
                b1.u[v]     = Vw[(16 + lm) * (NTOK / 2) + kc * 16 + 4*half + v];
                b0.u[4 + v] = Vw[(lm     ) * (NTOK / 2) + kc * 16 + 8 + 4*half + v];
                b1.u[4 + v] = Vw[(16 + lm) * (NTOK / 2) + kc * 16 + 8 + 4*half + v];
            }
            o0 = __builtin_amdgcn_wmma_f32_16x16x32_f16(
                false, ap.h, false, b0.h, (short)0, o0, false, false);
            o1 = __builtin_amdgcn_wmma_f32_16x16x32_f16(
                false, ap.h, false, b1.h, (short)0, o1, false, false);
        }

        // ---- store output; window-reverse + unshift folded into coords ----
#pragma unroll
        for (int v = 0; v < 8; ++v) {
            const int r  = r0 + v + 8 * half;
            const int iy = r >> 4, ix = r & 15;
            const int y  = (wy * 16 + iy + 8) & 63;
            const int x  = (wx * 16 + ix + 8) & 63;
            float* dst = out + ((size_t)b * 4096 + (size_t)y * 64 + x) * 192 + h * HD;
            dst[lm]      = o0[v];
            dst[16 + lm] = o1[v];
        }
    }
}

// ---------------------------------------------------------------------------
// inputs (setup_inputs order): qkv, table, mask, logit_scale, w1, b1, w2,
//                              index, H, W
// ---------------------------------------------------------------------------
extern "C" void kernel_launch(void* const* d_in, const int* in_sizes, int n_in,
                              void* d_out, int out_size, void* d_ws, size_t ws_size,
                              hipStream_t stream)
{
    const float* qkv   = (const float*)d_in[0];
    const float* table = (const float*)d_in[1];
    const float* maskg = (const float*)d_in[2];
    const float* ls    = (const float*)d_in[3];
    const float* w1    = (const float*)d_in[4];
    const float* b1    = (const float*)d_in[5];
    const float* w2    = (const float*)d_in[6];
    const int*   index = (const int*)d_in[7];
    float* out = (float*)d_out;

    (void)in_sizes; (void)n_in; (void)out_size;

    // workspace layout: [ biasTab: 6*961 f32 (rounded to 256B) | fused f16 table ]
    const size_t biasBytes  = ((size_t)NH * 961 * sizeof(float) + 255) & ~(size_t)255;
    const size_t fusedElems = (size_t)NH * NWIN * NTOK * NTOK;          // 6.29M
    const size_t needFused  = biasBytes + fusedElems * sizeof(f16);     // ~12.6 MB
    float* biasTab = (float*)d_ws;
    f16*   fused   = (f16*)((char*)d_ws + biasBytes);
    const bool useFused = (ws_size >= needFused);

    cpb_bias_kernel<<<(961 * NH + 255) / 256, 256, 0, stream>>>(table, w1, b1, w2, biasTab);

    const size_t shmem = (size_t)(NTOK * HD * 3 + 8 * 16 * NTOK) * sizeof(f16); // 112 KB
    const int nblocks = 256 * NH;                  // 1536 (window, head) problems

    if (useFused) {
        fuse_bias_mask_kernel<<<(int)((fusedElems + 255) / 256), 256, 0, stream>>>(
            index, maskg, biasTab, fused);
        swin_attn_kernel<true><<<nblocks, 256, shmem, stream>>>(
            qkv, maskg, ls, index, biasTab, fused, out);
    } else {
        swin_attn_kernel<false><<<nblocks, 256, shmem, stream>>>(
            qkv, maskg, ls, index, biasTab, fused, out);
    }
}
